// Attention_57080115364834
// MI455X (gfx1250) — compile-verified, matrix operand order checked by reading, not run on hardware
//
#include <hip/hip_runtime.h>

// Attention B=4, C=512, N=4096. All GEMMs on v_wmma_f32_16x16x32_bf16.
// Round-3 changes (flash kernel only):
//  - Q block (32x512 bf16) staged once in LDS; S-phase A-fragments now come
//    from ds_load_b128 instead of re-reading global 32 times per block.
//  - global_prefetch of next key-block K/V lines overlaps memory with WMMA.
//
// Workspace (needs ~82 MB):
//   Tb  : bf16 [B][N][C]  transposed tokens (x is [B][C][N])
//   Qb, Kb : bf16 [B][N][C]
//   Vt  : bf16 [B][C][N]  (transposed so P@V B-fragments are contiguous)
//   Ob  : bf16 [B][N][C]
//   Wqb/Wkb/Wvb/Wob : bf16 [C][C]

#define NB 4
#define CC 512
#define NN 4096

typedef __attribute__((ext_vector_type(16))) __bf16   v16bf;
typedef __attribute__((ext_vector_type(8)))  float    v8f;
typedef __attribute__((ext_vector_type(8)))  unsigned int v8u;
typedef unsigned int   u32;
typedef unsigned short u16;

__device__ __forceinline__ u16 cvt_bf16(float f) {
    return __builtin_bit_cast(u16, static_cast<__bf16>(f));
}

__device__ __forceinline__ v8f wmma_bf16(v16bf a, v16bf b, v8f c) {
    return __builtin_amdgcn_wmma_f32_16x16x32_bf16(false, a, false, b,
                                                   (short)0, c, false, false);
}

// 16-half A/B fragment from bf16 memory with contiguous K.
// kc = 0 (lanes 0-15) or 8 (lanes 16-31); per ISA 7.12.2 the lane holds
// K{kc..kc+7} in u32[0..3] and K{kc+16..kc+23} in u32[4..7].
__device__ __forceinline__ v16bf load_frag(const u16* p, int kc) {
    const uint4 lo = *(const uint4*)(p + kc);
    const uint4 hi = *(const uint4*)(p + kc + 16);
    v8u v;
    v[0] = lo.x; v[1] = lo.y; v[2] = lo.z; v[3] = lo.w;
    v[4] = hi.x; v[5] = hi.y; v[6] = hi.z; v[7] = hi.w;
    return __builtin_bit_cast(v16bf, v);
}

// ---------------------------------------------------------------------------
// Prep A: x [B][C][N] fp32  ->  Tb [B][N][C] bf16  (transpose + convert).
// ---------------------------------------------------------------------------
__global__ __launch_bounds__(256) void transpose_cvt_kernel(
    const float* __restrict__ x, u16* __restrict__ Tb)
{
    __shared__ u16 tl[64][72];                 // row stride 144B (16B multiple)
    const int n0 = blockIdx.x * 64, c0 = blockIdx.y * 64, b = blockIdx.z;
    const int t = threadIdx.x;

    for (int i = 0; i < 4; ++i) {
        const int idx = t + 256 * i;           // 0..1023
        const int c   = idx >> 4;              // 0..63
        const int n4  = idx & 15;              // 0..15  (x4 tokens)
        const float4 v = *(const float4*)(x + (size_t)(b * CC + c0 + c) * NN
                                            + n0 + n4 * 4);
        tl[n4 * 4 + 0][c] = cvt_bf16(v.x);
        tl[n4 * 4 + 1][c] = cvt_bf16(v.y);
        tl[n4 * 4 + 2][c] = cvt_bf16(v.z);
        tl[n4 * 4 + 3][c] = cvt_bf16(v.w);
    }
    __syncthreads();
    for (int i = 0; i < 2; ++i) {
        const int idx = t + 256 * i;           // 0..511
        const int n   = idx >> 3;              // 0..63
        const int cc8 = idx & 7;               // 8-element chunk of c
        const uint4 v = *(const uint4*)(&tl[n][cc8 * 8]);
        *(uint4*)(Tb + (size_t)(b * NN + n0 + n) * CC + c0 + cc8 * 8) = v;
    }
}

// ---------------------------------------------------------------------------
// Prep B: elementwise fp32 -> bf16 (weights).
// ---------------------------------------------------------------------------
__global__ __launch_bounds__(256) void cvt_w_kernel(
    const float* __restrict__ src, u16* __restrict__ dst)
{
    const int i = (blockIdx.x * 256 + threadIdx.x) * 8;
    const float4 a = *(const float4*)(src + i);
    const float4 c = *(const float4*)(src + i + 4);
    uint4 v;
    v.x = (u32)cvt_bf16(a.x) | ((u32)cvt_bf16(a.y) << 16);
    v.y = (u32)cvt_bf16(a.z) | ((u32)cvt_bf16(a.w) << 16);
    v.z = (u32)cvt_bf16(c.x) | ((u32)cvt_bf16(c.y) << 16);
    v.w = (u32)cvt_bf16(c.z) | ((u32)cvt_bf16(c.w) << 16);
    *(uint4*)(dst + i) = v;
}

// ---------------------------------------------------------------------------
// QKV projection GEMM, pure bf16 operands from global.
// Block 128x128, 8 waves in 4x2 (each wave 32 rows x 64 cols).
// ---------------------------------------------------------------------------
__global__ __launch_bounds__(256) void gemm_qkv_kernel(
    const u16* __restrict__ Tb, const u16* __restrict__ Wb,
    const float* __restrict__ bias, u16* __restrict__ out, int transpose_out)
{
    const int n0 = blockIdx.x * 128, co0 = blockIdx.y * 128, b = blockIdx.z;
    const int lane = threadIdx.x & 31, wv = threadIdx.x >> 5;
    const int wr = wv & 3, wc = wv >> 2;
    const int ln = lane & 15, hl = lane >> 4, kc = hl ? 8 : 0;

    const u16* arow0 = Tb + (size_t)(b * NN + n0 + wr * 32 + ln) * CC;
    const u16* arow1 = arow0 + (size_t)16 * CC;
    const u16* brow[4];
    for (int cc = 0; cc < 4; ++cc)
        brow[cc] = Wb + (size_t)(co0 + wc * 64 + cc * 16 + ln) * CC;

    v8f acc[2][4];
    const v8f vzero = {0.f, 0.f, 0.f, 0.f, 0.f, 0.f, 0.f, 0.f};
    for (int i = 0; i < 2; ++i)
        for (int j = 0; j < 4; ++j) acc[i][j] = vzero;

    for (int k0 = 0; k0 < CC; k0 += 32) {
        const v16bf a0 = load_frag(arow0 + k0, kc);
        const v16bf a1 = load_frag(arow1 + k0, kc);
        for (int cc = 0; cc < 4; ++cc) {
            const v16bf bf = load_frag(brow[cc] + k0, kc);
            acc[0][cc] = wmma_bf16(a0, bf, acc[0][cc]);
            acc[1][cc] = wmma_bf16(a1, bf, acc[1][cc]);
        }
    }

    for (int tr = 0; tr < 2; ++tr)
        for (int cc = 0; cc < 4; ++cc) {
            const int co = co0 + wc * 64 + cc * 16 + ln;
            const float bv = bias[co];
            for (int r = 0; r < 8; ++r) {
                const int n = n0 + wr * 32 + tr * 16 + r + 8 * hl;
                const u16 val = cvt_bf16(acc[tr][cc][r] + bv);
                const size_t addr = transpose_out
                    ? (size_t)(b * CC + co) * NN + n
                    : (size_t)(b * NN + n) * CC + co;
                out[addr] = val;
            }
        }
}

// ---------------------------------------------------------------------------
// Flash attention: Br=32 queries/block, Bc=128 keys/iter, 8 waves.
// Wave w computes score cols [16w,16w+16) and owns O d-slice [64w,64w+64).
// Q staged once in LDS; K/V stream from global with prefetch.
// ---------------------------------------------------------------------------
__global__ __launch_bounds__(256) void flash_attn_kernel(
    const u16* __restrict__ Qb, const u16* __restrict__ Kb,
    const u16* __restrict__ Vt, u16* __restrict__ Ob)
{
    __shared__ u16   Qld[32][520];             // 1040B row stride (16B multiple)
    __shared__ float Sld[32][132];
    __shared__ u16   Pld[32][136];             // 272B row stride (16B multiple)
    __shared__ float red[32][9];
    __shared__ float mrun[32], lrun[32], fac[32];

    const int b = blockIdx.y, qn0 = blockIdx.x * 32;
    const int t = threadIdx.x, lane = t & 31, wv = t >> 5;
    const int ln = lane & 15, hl = lane >> 4, kc = hl ? 8 : 0;
    const int rrow = t >> 3, rseg = t & 7;     // 8 threads per score row
    const float scale = 0.044194173824159216f; // 1/sqrt(512)

    // ---- stage Q block (32x512 bf16 = 32 KB) into LDS once ----
    for (int i = 0; i < 8; ++i) {
        const int idx = t + 256 * i;           // 0..2047
        const int row = idx >> 6;              // 0..31
        const int ch  = idx & 63;              // 8-element chunk
        const uint4 v = *(const uint4*)(Qb + (size_t)(b * NN + qn0 + row) * CC
                                           + ch * 8);
        *(uint4*)(&Qld[row][ch * 8]) = v;
    }

    v8f o[2][4];
    const v8f vzero = {0.f, 0.f, 0.f, 0.f, 0.f, 0.f, 0.f, 0.f};
    for (int i = 0; i < 2; ++i)
        for (int j = 0; j < 4; ++j) o[i][j] = vzero;

    if (t < 32) { mrun[t] = -1e30f; lrun[t] = 0.f; }
    __syncthreads();

    const u16* qrow0 = &Qld[ln][0];
    const u16* qrow1 = &Qld[16 + ln][0];

    for (int kb = 0; kb < NN; kb += 128) {
        // ---- prefetch next key block's K and V lines (global_prefetch_b8) ----
        if (kb + 128 < NN) {
            const u16* knext = Kb + (size_t)(b * NN + kb + 128 + wv * 16 + ln) * CC;
            __builtin_prefetch(knext, 0, 1);
            __builtin_prefetch(knext + 256, 0, 1);
            const u16* vnext = Vt + (size_t)(b * CC + wv * 64 + lane * 2) * NN
                                  + kb + 128;
            __builtin_prefetch(vnext, 0, 1);
        }

        // ---- S = Q @ K^T  (Q from LDS, K from global) ----
        v8f s0 = vzero, s1 = vzero;
        const u16* krow = Kb + (size_t)(b * NN + kb + wv * 16 + ln) * CC;
        for (int d0 = 0; d0 < CC; d0 += 32) {
            const v16bf bk = load_frag(krow + d0, kc);
            s0 = wmma_bf16(load_frag(qrow0 + d0, kc), bk, s0);
            s1 = wmma_bf16(load_frag(qrow1 + d0, kc), bk, s1);
        }
        const int col = wv * 16 + ln;
        for (int r = 0; r < 8; ++r) {
            Sld[r + 8 * hl][col]      = s0[r] * scale;
            Sld[16 + r + 8 * hl][col] = s1[r] * scale;
        }
        __syncthreads();

        // ---- online softmax stats (8 threads per row) ----
        float mx = -1e30f;
        for (int j = 0; j < 16; ++j) mx = fmaxf(mx, Sld[rrow][rseg * 16 + j]);
        red[rrow][rseg] = mx;
        __syncthreads();
        if (rseg == 0) {
            float mkb = red[rrow][0];
            for (int j = 1; j < 8; ++j) mkb = fmaxf(mkb, red[rrow][j]);
            const float m_old = mrun[rrow];
            const float mnew  = fmaxf(m_old, mkb);
            fac[rrow]  = __expf(m_old - mnew);
            mrun[rrow] = mnew;
        }
        __syncthreads();
        {
            const float mnew = mrun[rrow];
            float sum = 0.f;
            for (int j = 0; j < 16; ++j) {
                const float p = __expf(Sld[rrow][rseg * 16 + j] - mnew);
                Pld[rrow][rseg * 16 + j] = cvt_bf16(p);
                sum += p;
            }
            red[rrow][rseg] = sum;
        }
        __syncthreads();
        if (rseg == 0) {
            float sum = red[rrow][0];
            for (int j = 1; j < 8; ++j) sum += red[rrow][j];
            lrun[rrow] = lrun[rrow] * fac[rrow] + sum;
        }

        // ---- rescale running O ----
        for (int tr = 0; tr < 2; ++tr)
            for (int r = 0; r < 8; ++r) {
                const float f = fac[tr * 16 + r + 8 * hl];
                for (int cc = 0; cc < 4; ++cc) o[tr][cc][r] *= f;
            }

        // ---- O += P @ V  (P from LDS, V^T from global) ----
        for (int kk0 = 0; kk0 < 128; kk0 += 32) {
            const v16bf p0 = load_frag(&Pld[ln][kk0], kc);
            const v16bf p1 = load_frag(&Pld[16 + ln][kk0], kc);
            for (int cc = 0; cc < 4; ++cc) {
                const int d = wv * 64 + cc * 16 + ln;
                const u16* vrow = Vt + (size_t)(b * CC + d) * NN + kb + kk0;
                const v16bf bv = load_frag(vrow, kc);
                o[0][cc] = wmma_bf16(p0, bv, o[0][cc]);
                o[1][cc] = wmma_bf16(p1, bv, o[1][cc]);
            }
        }
        __syncthreads();    // Sld/Pld/lrun safe before next iteration
    }

    for (int tr = 0; tr < 2; ++tr)
        for (int r = 0; r < 8; ++r) {
            const float inv = 1.f / lrun[tr * 16 + r + 8 * hl];
            const int n = qn0 + tr * 16 + r + 8 * hl;
            for (int cc = 0; cc < 4; ++cc) {
                const int d = wv * 64 + cc * 16 + ln;
                Ob[(size_t)(b * NN + n) * CC + d] = cvt_bf16(o[tr][cc][r] * inv);
            }
        }
}

// ---------------------------------------------------------------------------
// Output projection: fp32 result in the reference's no-permute layout
// d_out[b*N*C + n*C + co].
// ---------------------------------------------------------------------------
__global__ __launch_bounds__(256) void out_proj_kernel(
    const u16* __restrict__ Ob, const u16* __restrict__ Wob,
    const float* __restrict__ bo, float* __restrict__ out)
{
    const int n0 = blockIdx.x * 128, co0 = blockIdx.y * 128, b = blockIdx.z;
    const int lane = threadIdx.x & 31, wv = threadIdx.x >> 5;
    const int wr = wv & 3, wc = wv >> 2;
    const int ln = lane & 15, hl = lane >> 4, kc = hl ? 8 : 0;

    const u16* arow0 = Ob + (size_t)(b * NN + n0 + wr * 32 + ln) * CC;
    const u16* arow1 = arow0 + (size_t)16 * CC;
    const u16* brow[4];
    for (int cc = 0; cc < 4; ++cc)
        brow[cc] = Wob + (size_t)(co0 + wc * 64 + cc * 16 + ln) * CC;

    v8f acc[2][4];
    const v8f vzero = {0.f, 0.f, 0.f, 0.f, 0.f, 0.f, 0.f, 0.f};
    for (int i = 0; i < 2; ++i)
        for (int j = 0; j < 4; ++j) acc[i][j] = vzero;

    for (int k0 = 0; k0 < CC; k0 += 32) {
        const v16bf a0 = load_frag(arow0 + k0, kc);
        const v16bf a1 = load_frag(arow1 + k0, kc);
        for (int cc = 0; cc < 4; ++cc) {
            const v16bf bf = load_frag(brow[cc] + k0, kc);
            acc[0][cc] = wmma_bf16(a0, bf, acc[0][cc]);
            acc[1][cc] = wmma_bf16(a1, bf, acc[1][cc]);
        }
    }

    for (int tr = 0; tr < 2; ++tr)
        for (int cc = 0; cc < 4; ++cc) {
            const int co = co0 + wc * 64 + cc * 16 + ln;
            const float bb = bo[co];
            for (int r = 0; r < 8; ++r) {
                const int n = n0 + wr * 32 + tr * 16 + r + 8 * hl;
                out[(size_t)(b * NN + n) * CC + co] = acc[tr][cc][r] + bb;
            }
        }
}

extern "C" void kernel_launch(void* const* d_in, const int* in_sizes, int n_in,
                              void* d_out, int out_size, void* d_ws, size_t ws_size,
                              hipStream_t stream)
{
    const float* x  = (const float*)d_in[0];
    const float* Wq = (const float*)d_in[1];
    const float* bq = (const float*)d_in[2];
    const float* Wk = (const float*)d_in[3];
    const float* bk = (const float*)d_in[4];
    const float* Wv = (const float*)d_in[5];
    const float* bv = (const float*)d_in[6];
    const float* Wo = (const float*)d_in[7];
    const float* bo = (const float*)d_in[8];

    const size_t tok = (size_t)NB * NN * CC;   // elements per token-sized buffer
    const size_t wsz = (size_t)CC * CC;        // elements per weight matrix
    u16* Tb  = (u16*)d_ws;
    u16* Qb  = Tb + tok;
    u16* Kb  = Qb + tok;
    u16* Vt  = Kb + tok;                       // [B][C][N]
    u16* Ob  = Vt + tok;
    u16* Wqb = Ob + tok;
    u16* Wkb = Wqb + wsz;
    u16* Wvb = Wkb + wsz;
    u16* Wob = Wvb + wsz;

    transpose_cvt_kernel<<<dim3(NN / 64, CC / 64, NB), 256, 0, stream>>>(x, Tb);
    const int wblocks = (int)(wsz / (256 * 8));
    cvt_w_kernel<<<wblocks, 256, 0, stream>>>(Wq, Wqb);
    cvt_w_kernel<<<wblocks, 256, 0, stream>>>(Wk, Wkb);
    cvt_w_kernel<<<wblocks, 256, 0, stream>>>(Wv, Wvb);
    cvt_w_kernel<<<wblocks, 256, 0, stream>>>(Wo, Wob);

    const dim3 gProj(NN / 128, CC / 128, NB);
    gemm_qkv_kernel<<<gProj, 256, 0, stream>>>(Tb, Wqb, bq, Qb, 0);
    gemm_qkv_kernel<<<gProj, 256, 0, stream>>>(Tb, Wkb, bk, Kb, 0);
    gemm_qkv_kernel<<<gProj, 256, 0, stream>>>(Tb, Wvb, bv, Vt, 1);

    flash_attn_kernel<<<dim3(NN / 32, NB), 256, 0, stream>>>(Qb, Kb, Vt, Ob);

    out_proj_kernel<<<gProj, 256, 0, stream>>>(Ob, Wob, bo, (float*)d_out);
}
